// NonLocalDenoiser_41291815584231
// MI455X (gfx1250) — matrix operand order repaired and checked by reading, not compiled, not run on hardware
//
#include <hip/hip_runtime.h>
#include <hip/hip_bf16.h>

// ---------------------------------------------------------------------------
// NonLocalDenoiser forward for MI455X (gfx1250, wave32, WMMA).
// Core op: verhor  Y = Wh * (X * Wv^T) + b  done with v_wmma_f32_16x16x32_f16,
// one patch per wave, LDS-staged f16 tiles (ds_load_b128 fragment gathers,
// async global->LDS staging when available). Activations kept in f16 between
// layers to halve HBM traffic (23.3 TB/s chip; whole net ~141 GFLOP).
// ---------------------------------------------------------------------------

typedef __attribute__((ext_vector_type(16))) _Float16 v16h;
typedef __attribute__((ext_vector_type(8)))  float    v8f;
typedef __attribute__((ext_vector_type(4)))  unsigned v4u;
typedef __attribute__((ext_vector_type(4)))  int      v4i;

static constexpr int IPN   = 32768;   // I*P = 2*16384
static constexpr int PIMG  = 16384;   // patches per image (128*128)
static constexpr float BN_EPS = 1e-5f;

__host__ __device__ constexpr int r16i(int x) { return (x + 15) & ~15; }
__host__ __device__ constexpr int r32i(int x) { return (x + 31) & ~31; }

// LDS halves layout: [ WvT: K1*NP | WhA: MP2*K2 | per-wave X: MP1*K1 | per-wave T: K2*NP ]
__host__ __device__ constexpr size_t vh_lds_bytes(int hi, int ho, int vi, int vo) {
    int MP1 = r16i(hi), K1 = r32i(vi), NP = r16i(vo), K2 = r32i(hi), MP2 = r16i(ho);
    return (size_t)2 * (size_t)(K1 * NP + MP2 * K2 + 8 * (MP1 * K1 + K2 * NP));
}

#if defined(__HIP_DEVICE_COMPILE__) && \
    __has_builtin(__builtin_amdgcn_global_load_async_to_lds_b128) && \
    __has_builtin(__builtin_amdgcn_s_wait_asynccnt)
#define USE_ASYNC_LDS 1
#else
#define USE_ASYNC_LDS 0
#endif

// --- Fragment loaders (ISA 7.12.2 layouts), data staged in LDS as f16 ------
// All tiles are 16-byte aligned in LDS -> ds_load_b128.

// A: 16x32 f16 tile. lane m = lane&15 ; K base kb = (lane>>4)*8 ;
// halves 0..7 cover K = kb+0..7 and kb+16..23  (two contiguous 16B groups).
__device__ __forceinline__ v16h load_a_frag(const _Float16* base, int ld, int mt, int kt, int lane) {
    int m  = lane & 15;
    int kb = (lane >> 4) * 8;
    const _Float16* rp = base + (size_t)(mt * 16 + m) * ld + kt * 32;
    union { v16h v; v4u q[2]; } fr;
    fr.q[0] = *(const v4u*)(rp + kb);
    fr.q[1] = *(const v4u*)(rp + kb + 16);
    return fr.v;
}

// B: 32x16 f16 tile, [K][N] row-major in LDS. lane -> N = lane&15 ; K base kb = (lane>>4)*8 ;
// halves 0..7 cover K = kb+0..7 and kb+16..23 (strided gather).
__device__ __forceinline__ v16h load_b_frag(const _Float16* base, int ld, int kt, int nt, int lane) {
    int n  = lane & 15;
    int kb = (lane >> 4) * 8;
    const _Float16* cp = base + (size_t)(kt * 32) * ld + nt * 16 + n;
    v16h v;
#pragma unroll
    for (int s = 0; s < 8; ++s) {
        v[s]     = cp[(size_t)(kb + s) * ld];
        v[8 + s] = cp[(size_t)(kb + 16 + s) * ld];
    }
    return v;
}

// --- Fused verhor kernel ----------------------------------------------------
// scale_mode: 0 none, 1 multiply by scale[patch], 2 divide by scale[patch]
template <int H_IN, int H_OUT, int V_IN, int V_OUT, bool IN_HALF, bool OUT_HALF, bool RELU>
__global__ __launch_bounds__(256) void verhor_kernel(
    const void* __restrict__ xin, long in_ps,
    const float* __restrict__ Wv, const float* __restrict__ Wh, const float* __restrict__ bias,
    void* __restrict__ out, long out_ps, int out_rs,
    const float* __restrict__ scale, int scale_mode, int n_patches)
{
    constexpr int MP1 = r16i(H_IN),  MT1 = MP1 / 16;
    constexpr int K1  = r32i(V_IN),  KT1 = K1 / 32;
    constexpr int NP  = r16i(V_OUT), NT  = NP / 16;
    constexpr int K2  = r32i(H_IN),  KT2 = K2 / 32;
    constexpr int MP2 = r16i(H_OUT), MT2 = MP2 / 16;

    extern __shared__ _Float16 smem[];
    _Float16* WvT = smem;                               // [K1][NP]
    _Float16* WhA = WvT + K1 * NP;                      // [MP2][K2]
    _Float16* Xw  = WhA + MP2 * K2;                     // 8 waves * [MP1][K1]
    _Float16* Tw  = Xw + 8 * MP1 * K1;                  // 8 waves * [K2][NP]

    const int tid  = threadIdx.x;
    const int wave = tid >> 5;
    const int lane = tid & 31;

    // Cooperative weight staging: WvT[k][n] = Wv[n][k]  (Wv is [v_out, v_in])
    for (int i = tid; i < K1 * NP; i += 256) {
        int k = i / NP, n = i % NP;
        WvT[i] = (_Float16)((k < V_IN && n < V_OUT) ? Wv[n * V_IN + k] : 0.f);
    }
    // WhA[m][k] = Wh[m][k]  (Wh is [h_out, h_in]), zero padded
    for (int i = tid; i < MP2 * K2; i += 256) {
        int m = i / K2, k = i % K2;
        WhA[i] = (_Float16)((m < H_OUT && k < H_IN) ? Wh[m * H_IN + k] : 0.f);
    }
    __syncthreads();

    const int patch = blockIdx.x * 8 + wave;
    if (patch >= n_patches) return;           // uniform per wave -> EXEC stays full

    _Float16* X = Xw + wave * (MP1 * K1);
    _Float16* T = Tw + wave * (K2 * NP);

    // Zero only the padding of X (pad columns + pad rows); data region is
    // written below.  T fully zeroed (stage1 overwrites rows < MP1; DS ops of
    // one wave stay in order).
    for (int i = lane; i < H_IN * (K1 - V_IN); i += 32) {
        int r = i / (K1 - V_IN), c = V_IN + i % (K1 - V_IN);
        X[r * K1 + c] = (_Float16)0.f;
    }
    for (int i = lane; i < (MP1 - H_IN) * K1; i += 32) X[H_IN * K1 + i] = (_Float16)0.f;
    for (int i = lane; i < K2 * NP; i += 32) T[i] = (_Float16)0.f;

    // Load & convert this patch's X into LDS
    if (IN_HALF) {
        const _Float16* xp = (const _Float16*)xin + (long)patch * in_ps;
#if USE_ASYNC_LDS
        // f16 rows are 16B-aligned for every config used (V_IN multiple of 8,
        // patch strides 2240/8960 B): stream them with the gfx1250 async path.
        if ((V_IN % 8) == 0) {
            constexpr int CPR = (V_IN % 8 == 0) ? V_IN / 8 : 1;  // 16B chunks per row
            for (int i = lane; i < H_IN * CPR; i += 32) {
                int r = i / CPR, c = i % CPR;
                __builtin_amdgcn_global_load_async_to_lds_b128(
                    (__attribute__((address_space(1))) v4i*)(xp + (size_t)r * V_IN + c * 8),
                    (__attribute__((address_space(3))) v4i*)(X + (size_t)r * K1 + c * 8),
                    0, 0);
            }
            __builtin_amdgcn_s_wait_asynccnt(0);
        } else
#endif
        {
            __builtin_prefetch(xp, 0, 0);
            for (int i = lane; i < H_IN * V_IN; i += 32) {
                int r = i / V_IN, c = i % V_IN;
                X[r * K1 + c] = xp[i];
            }
        }
    } else {
        const float* xp = (const float*)xin + (long)patch * in_ps;
        __builtin_prefetch(xp, 0, 0);
        for (int i = lane; i < H_IN * V_IN; i += 32) {
            int r = i / V_IN, c = i % V_IN;
            X[r * K1 + c] = (_Float16)xp[i];
        }
    }

    // Stage 1: T = X * Wv^T   (M=H_IN, N=V_OUT, K=V_IN)
#pragma unroll 1
    for (int mt = 0; mt < MT1; ++mt) {
        v16h af[KT1];
#pragma unroll
        for (int kt = 0; kt < KT1; ++kt) af[kt] = load_a_frag(X, K1, mt, kt, lane);
#pragma unroll
        for (int nt = 0; nt < NT; ++nt) {
            v16h bf[KT1];
#pragma unroll
            for (int kt = 0; kt < KT1; ++kt) bf[kt] = load_b_frag(WvT, NP, kt, nt, lane);
            v8f acc = {};
#pragma unroll
            for (int kt = 0; kt < KT1; ++kt)
                acc = __builtin_amdgcn_wmma_f32_16x16x32_f16(false, af[kt], false, bf[kt],
                                                             (short)0, acc, false, false);
            int col = nt * 16 + (lane & 15);
            int rb  = (lane >> 4) * 8;
#pragma unroll
            for (int r = 0; r < 8; ++r)
                T[(size_t)(mt * 16 + rb + r) * NP + col] = (_Float16)acc[r];
        }
    }

    // Stage 2: Y = Wh * T + b  (M=H_OUT, N=V_OUT, K=H_IN)
    float sc = 1.f;
    if (scale_mode) {
        float w = scale[patch];
        sc = (scale_mode == 2) ? (1.f / w) : w;
    }
#pragma unroll 1
    for (int mt = 0; mt < MT2; ++mt) {
        v16h af[KT2];
#pragma unroll
        for (int kt = 0; kt < KT2; ++kt) af[kt] = load_a_frag(WhA, K2, mt, kt, lane);
#pragma unroll
        for (int nt = 0; nt < NT; ++nt) {
            v16h bf[KT2];
#pragma unroll
            for (int kt = 0; kt < KT2; ++kt) bf[kt] = load_b_frag(T, NP, kt, nt, lane);
            v8f acc = {};
#pragma unroll
            for (int kt = 0; kt < KT2; ++kt)
                acc = __builtin_amdgcn_wmma_f32_16x16x32_f16(false, af[kt], false, bf[kt],
                                                             (short)0, acc, false, false);
            int col = nt * 16 + (lane & 15);
            int rb  = (lane >> 4) * 8;
#pragma unroll
            for (int r = 0; r < 8; ++r) {
                int row = mt * 16 + rb + r;
                if (row < H_OUT && col < V_OUT) {
                    float v = acc[r] + bias[row * V_OUT + col];
                    if (scale_mode) v *= sc;
                    if (RELU) v = fmaxf(v, 0.f);
                    long o = (long)patch * out_ps + (long)row * out_rs + col;
                    if (OUT_HALF) ((_Float16*)out)[o] = (_Float16)v;
                    else          ((float*)out)[o]    = v;
                }
            }
        }
    }
}

// --- BatchNorm stats: per-channel sum / sumsq over (patches, V) -------------
__global__ void zero_kernel(float* p, int n) {
    int i = blockIdx.x * blockDim.x + threadIdx.x;
    if (i < n) p[i] = 0.f;
}

__global__ void bn_stats_kernel(const float* __restrict__ u, float* __restrict__ stats,
                                int C, int V, int npatch)
{
    int c    = blockIdx.x % C;
    int blk  = blockIdx.x / C;
    int nblk = gridDim.x / C;
    float s = 0.f, q = 0.f;
    long per = (long)npatch * V;
    for (long t = (long)blk * blockDim.x + threadIdx.x; t < per; t += (long)nblk * blockDim.x) {
        long p = t / V; int v = (int)(t % V);
        float x = u[((long)p * C + c) * V + v];
        s += x; q += x * x;
    }
    __shared__ float ss[256], sq[256];
    ss[threadIdx.x] = s; sq[threadIdx.x] = q;
    __syncthreads();
    for (int k = 128; k > 0; k >>= 1) {
        if (threadIdx.x < k) { ss[threadIdx.x] += ss[threadIdx.x + k]; sq[threadIdx.x] += sq[threadIdx.x + k]; }
        __syncthreads();
    }
    if (threadIdx.x == 0) {
        atomicAdd(&stats[c], ss[0]);
        atomicAdd(&stats[C + c], sq[0]);
    }
}

// --- BN apply + ReLU + optional residual mix: out = a*x + (1-a)*relu(bn(u)) -
__global__ void bn_mix_kernel(const _Float16* __restrict__ xin, long x_ps, int x_rs,
                              const float* __restrict__ u, const float* __restrict__ stats,
                              const float* __restrict__ g, const float* __restrict__ b,
                              const float* __restrict__ alpha,
                              _Float16* __restrict__ out, long o_ps, int o_rs,
                              int C, int V, int npatch, float invN)
{
    long total = (long)npatch * C * V;
    long stride = (long)gridDim.x * blockDim.x;
    for (long idx = (long)blockIdx.x * blockDim.x + threadIdx.x; idx < total; idx += stride) {
        int  v = (int)(idx % V);
        int  c = (int)((idx / V) % C);
        long p = idx / ((long)C * V);
        float mu  = stats[c] * invN;
        float var = stats[C + c] * invN - mu * mu;
        float y = (u[idx] - mu) * rsqrtf(var + BN_EPS) * g[c] + b[c];
        y = fmaxf(y, 0.f);
        if (xin) {
            float a = *alpha;
            y = a * (float)xin[p * x_ps + (long)c * x_rs + v] + (1.f - a) * y;
        }
        out[p * o_ps + (long)c * o_rs + v] = (_Float16)y;
    }
}

// --- fold (as gather) + overlap-count normalize -----------------------------
// y: [I, P, 75] f32 ; f: [I, 3, H, W] f32 ; nH = nW = 128
__global__ void fold_norm_kernel(const float* __restrict__ y, float* __restrict__ f,
                                 int Himg, int Wimg, int dil, int n)
{
    int idx = blockIdx.x * blockDim.x + threadIdx.x;
    if (idx >= n) return;
    int hw = Himg * Wimg;
    int bc = idx / hw;           // i*3 + c
    int i  = bc / 3, c = bc % 3;
    int h  = (idx / Wimg) % Himg;
    int w  = idx % Wimg;
    float sum = 0.f; int cnt = 0;
#pragma unroll
    for (int ki = 0; ki < 5; ++ki) {
        int oi = h - ki * dil;
        if (oi < 0 || oi >= 128) continue;
#pragma unroll
        for (int kj = 0; kj < 5; ++kj) {
            int oj = w - kj * dil;
            if (oj < 0 || oj >= 128) continue;
            long gp = (long)i * PIMG + oi * 128 + oj;
            sum += y[gp * 75 + c * 25 + ki * 5 + kj];
            ++cnt;
        }
    }
    f[idx] = sum / (float)cnt;
}

// 3x3 separable [.25 .5 .25] smoother with reflect padding
__global__ void smooth_kernel(const float* __restrict__ f, float* __restrict__ o,
                              int Himg, int Wimg, int n)
{
    int idx = blockIdx.x * blockDim.x + threadIdx.x;
    if (idx >= n) return;
    int hw = Himg * Wimg;
    int bc = idx / hw;
    int h  = (idx / Wimg) % Himg;
    int w  = idx % Wimg;
    const float kw[3] = {0.25f, 0.5f, 0.25f};
    float acc = 0.f;
#pragma unroll
    for (int dh = -1; dh <= 1; ++dh) {
        int rh = h + dh;
        rh = rh < 0 ? -rh : (rh >= Himg ? 2 * Himg - 2 - rh : rh);
#pragma unroll
        for (int dw = -1; dw <= 1; ++dw) {
            int rw = w + dw;
            rw = rw < 0 ? -rw : (rw >= Wimg ? 2 * Wimg - 2 - rw : rw);
            acc += kw[dh + 1] * kw[dw + 1] * f[(long)bc * hw + rh * Wimg + rw];
        }
    }
    o[idx] = acc;
}

// unfold (gather) with optional per-patch multiply by weights1
__global__ void unfold_kernel(const float* __restrict__ f, float* __restrict__ y,
                              const float* __restrict__ w1, int Himg, int Wimg,
                              int dil, int n)
{
    int idx = blockIdx.x * blockDim.x + threadIdx.x;
    if (idx >= n) return;
    int  k  = idx % 75;
    long gp = idx / 75;                 // global patch 0..32767
    int  i  = (int)(gp / PIMG);
    int  p  = (int)(gp % PIMG);
    int oi = p >> 7, oj = p & 127;
    int c = k / 25, ki = (k % 25) / 5, kj = k % 5;
    float v = f[(((long)i * 3 + c) * Himg + oi + ki * dil) * Wimg + oj + kj * dil];
    if (w1) v *= w1[gp];
    y[idx] = v;
}

// ---------------------------------------------------------------------------
extern "C" void kernel_launch(void* const* d_in, const int* in_sizes, int n_in,
                              void* d_out, int out_size, void* d_ws, size_t ws_size,
                              hipStream_t stream)
{
    (void)in_sizes; (void)n_in; (void)out_size; (void)ws_size;
    auto fp = [&](int i) { return (const float*)d_in[i]; };
    // setup_inputs() insertion order
    const float* x0 = fp(0);
    const float* x1 = fp(1);
    const float* w1 = fp(2);
    const float *s0v0w = fp(3),  *s0v0h = fp(4),  *s0v0b = fp(5),  *s0a = fp(6),
                *s0v1w = fp(7),  *s0v1h = fp(8),  *s0v1b = fp(9),  *s0g = fp(10), *s0bb = fp(11);
    const float *s1v0w = fp(12), *s1v0h = fp(13), *s1v0b = fp(14), *s1a = fp(15),
                *s1v1w = fp(16), *s1v1h = fp(17), *s1v1b = fp(18), *s1g = fp(19), *s1bb = fp(20);
    const float *a0pw = fp(21), *a0ph = fp(22), *a0pb = fp(23);
    const float *a1pw = fp(24), *a1ph = fp(25), *a1pb = fp(26);
    const float *a0qw = fp(27), *a0qh = fp(28), *a0qb = fp(29);
    const float *a1qw = fp(30), *a1qh = fp(31), *a1qb = fp(32);
    const float *v2w = fp(33), *v2h = fp(34), *v2b = fp(35), *bn2g = fp(36), *bn2b = fp(37);
    const float *a2  = fp(38);
    const float *v3w = fp(39), *v3h = fp(40), *v3b = fp(41), *bn3g = fp(42), *bn3b = fp(43);
    const float *outw = fp(44), *outh = fp(45), *outb = fp(46);

    // workspace bump allocator
    size_t off = 0;
    auto alloc = [&](size_t bytes) {
        size_t o = (off + 255) & ~(size_t)255;
        off = o + bytes;
        return (void*)((char*)d_ws + o);
    };
    _Float16* A   = (_Float16*)alloc((size_t)IPN * 14 * 80 * 2);  // relu(verhor0)
    float*    U14 = (float*)   alloc((size_t)IPN * 14 * 80 * 4);  // pre-BN (sep)
    _Float16* CC  = (_Float16*)alloc((size_t)IPN * 56 * 80 * 2);  // concat / h2
    float*    U56 = (float*)   alloc((size_t)IPN * 56 * 80 * 4);  // pre-BN (v2/v3)
    _Float16* Hh  = (_Float16*)alloc((size_t)IPN * 56 * 80 * 2);  // h = relu(bn2)
    float*    P0  = (float*)   alloc((size_t)IPN * 75 * 4);       // agg scratch
    float*    F1  = (float*)   alloc((size_t)2 * 3 * 136 * 136 * 4);
    float*    F2  = (float*)   alloc((size_t)2 * 3 * 136 * 136 * 4);
    float*    ST  = (float*)   alloc(2 * 56 * 4);

    const int GG  = (IPN + 7) / 8;   // verhor grid (8 patches / block)
    const dim3 B256(256);
    const float invN = 1.f / ((float)IPN * 80.f);
    auto gr = [](long n) { return dim3((unsigned)((n + 255) / 256)); };

    // ---- sep_part1 for x0 ----
    verhor_kernel<14,14,75,80,false,true,true><<<GG,B256,vh_lds_bytes(14,14,75,80),stream>>>(
        x0, 14*75, s0v0w, s0v0h, s0v0b, A, 14*80, 80, nullptr, 0, IPN);
    verhor_kernel<14,14,80,80,true,false,false><<<GG,B256,vh_lds_bytes(14,14,80,80),stream>>>(
        A, 14*80, s0v1w, s0v1h, s0v1b, U14, 14*80, 80, nullptr, 0, IPN);
    zero_kernel<<<1,128,0,stream>>>(ST, 28);
    bn_stats_kernel<<<14*8,B256,0,stream>>>(U14, ST, 14, 80, IPN);
    bn_mix_kernel<<<4096,B256,0,stream>>>(A, 14*80, 80, U14, ST, s0g, s0bb, s0a,
                                          CC + 0*80, 56*80, 80, 14, 80, IPN, invN);
    // ---- sep_part1 for x1 ----
    verhor_kernel<14,14,75,80,false,true,true><<<GG,B256,vh_lds_bytes(14,14,75,80),stream>>>(
        x1, 14*75, s1v0w, s1v0h, s1v0b, A, 14*80, 80, nullptr, 0, IPN);
    verhor_kernel<14,14,80,80,true,false,false><<<GG,B256,vh_lds_bytes(14,14,80,80),stream>>>(
        A, 14*80, s1v1w, s1v1h, s1v1b, U14, 14*80, 80, nullptr, 0, IPN);
    zero_kernel<<<1,128,0,stream>>>(ST, 28);
    bn_stats_kernel<<<14*8,B256,0,stream>>>(U14, ST, 14, 80, IPN);
    bn_mix_kernel<<<4096,B256,0,stream>>>(A, 14*80, 80, U14, ST, s1g, s1bb, s1a,
                                          CC + 14*80, 56*80, 80, 14, 80, IPN, invN);

    // ---- y0: pre -> agg0 (dil=1, 132x132) -> post ----
    verhor_kernel<14,1,80,75,true,false,false><<<GG,B256,vh_lds_bytes(14,1,80,75),stream>>>(
        CC + 0*80, 56*80, a0pw, a0ph, a0pb, P0, 75, 75, nullptr, 0, IPN);
    fold_norm_kernel<<<gr(2L*3*132*132),B256,0,stream>>>(P0, F1, 132, 132, 1, 2*3*132*132);
    unfold_kernel<<<gr((long)IPN*75),B256,0,stream>>>(F1, P0, nullptr, 132, 132, 1, IPN*75);
    verhor_kernel<1,14,75,80,false,true,true><<<GG,B256,vh_lds_bytes(1,14,75,80),stream>>>(
        P0, 75, a0qw, a0qh, a0qb, CC + 28*80, 56*80, 80, nullptr, 0, IPN);

    // ---- y1: pre (/w1) -> agg1 (dil=2, 136x136, smoothed) -> (*w1) -> post ----
    verhor_kernel<14,1,80,75,true,false,false><<<GG,B256,vh_lds_bytes(14,1,80,75),stream>>>(
        CC + 14*80, 56*80, a1pw, a1ph, a1pb, P0, 75, 75, w1, 2, IPN);
    fold_norm_kernel<<<gr(2L*3*136*136),B256,0,stream>>>(P0, F1, 136, 136, 2, 2*3*136*136);
    smooth_kernel<<<gr(2L*3*136*136),B256,0,stream>>>(F1, F2, 136, 136, 2*3*136*136);
    unfold_kernel<<<gr((long)IPN*75),B256,0,stream>>>(F2, P0, w1, 136, 136, 2, IPN*75);
    verhor_kernel<1,14,75,80,false,true,true><<<GG,B256,vh_lds_bytes(1,14,75,80),stream>>>(
        P0, 75, a1qw, a1qh, a1qb, CC + 42*80, 56*80, 80, nullptr, 0, IPN);

    // ---- trunk: v2 + bn2 -> h ; v3 + bn3 -> g ; h2 = a2*h + (1-a2)*g ----
    verhor_kernel<56,56,80,80,true,false,false><<<GG,B256,vh_lds_bytes(56,56,80,80),stream>>>(
        CC, 56*80, v2w, v2h, v2b, U56, 56*80, 80, nullptr, 0, IPN);
    zero_kernel<<<1,128,0,stream>>>(ST, 112);
    bn_stats_kernel<<<56*8,B256,0,stream>>>(U56, ST, 56, 80, IPN);
    bn_mix_kernel<<<4096,B256,0,stream>>>(nullptr, 0, 0, U56, ST, bn2g, bn2b, nullptr,
                                          Hh, 56*80, 80, 56, 80, IPN, invN);
    verhor_kernel<56,56,80,80,true,false,false><<<GG,B256,vh_lds_bytes(56,56,80,80),stream>>>(
        Hh, 56*80, v3w, v3h, v3b, U56, 56*80, 80, nullptr, 0, IPN);
    zero_kernel<<<1,128,0,stream>>>(ST, 112);
    bn_stats_kernel<<<56*8,B256,0,stream>>>(U56, ST, 56, 80, IPN);
    bn_mix_kernel<<<4096,B256,0,stream>>>(Hh, 56*80, 80, U56, ST, bn3g, bn3b, a2,
                                          CC, 56*80, 80, 56, 80, IPN, invN);

    // ---- output layer: [56,80] -> [1,75] f32 into d_out ----
    verhor_kernel<56,1,80,75,true,false,false><<<GG,B256,vh_lds_bytes(56,1,80,75),stream>>>(
        CC, 56*80, outw, outh, outb, (float*)d_out, 75, 75, nullptr, 0, IPN);
}
